// BasicTransformerBlock_47141561041261
// MI455X (gfx1250) — compile-verified, matrix operand order checked by reading, not run on hardware
//
#include <hip/hip_runtime.h>

typedef __attribute__((ext_vector_type(16))) __bf16 v16bf;
typedef __attribute__((ext_vector_type(8)))  float  v8f;

union BF16Frag {
    v16bf v;
    unsigned int u[8];
    __bf16 h[16];
};

#define DIM     512
#define HEADS   8
#define D_HEAD  64
#define SEQ_N   4096
#define BATCH   2
#define CTX_M   256
#define CTX_DIM 768
#define FF_INNER 2048
#define ROWS    (BATCH * SEQ_N)   // 8192
#define CROWS   (BATCH * CTX_M)   // 512
#define ATTN_SCALE 0.125f         // 64^-0.5

// ---------------------------------------------------------------------------
// DPP16 16-lane butterfly reductions (lanes 0-15 and 16-31 reduce separately,
// which matches the WMMA C-layout half-rows). Pure VALU, no LDS round-trips.
// ---------------------------------------------------------------------------
#define DPP_QUAD_XOR1 0xB1   // quad_perm(1,0,3,2)
#define DPP_QUAD_XOR2 0x4E   // quad_perm(2,3,0,1)
#define DPP_ROW_HMIRR 0x141  // row_half_mirror (crosses 4-boundary after quad reduce)
#define DPP_ROW_MIRR  0x140  // row_mirror      (crosses 8-boundary after that)

template <int CTRL>
__device__ __forceinline__ float dpp_movf(float x) {
    int i = __builtin_amdgcn_update_dpp(0, __float_as_int(x), CTRL, 0xf, 0xf, true);
    return __int_as_float(i);
}

__device__ __forceinline__ float red_max16(float x) {
    x = fmaxf(x, dpp_movf<DPP_QUAD_XOR1>(x));
    x = fmaxf(x, dpp_movf<DPP_QUAD_XOR2>(x));
    x = fmaxf(x, dpp_movf<DPP_ROW_HMIRR>(x));
    x = fmaxf(x, dpp_movf<DPP_ROW_MIRR>(x));
    return x;
}
__device__ __forceinline__ float red_sum16(float x) {
    x += dpp_movf<DPP_QUAD_XOR1>(x);
    x += dpp_movf<DPP_QUAD_XOR2>(x);
    x += dpp_movf<DPP_ROW_HMIRR>(x);
    x += dpp_movf<DPP_ROW_MIRR>(x);
    return x;
}

// ---------------------------------------------------------------------------
// fp32 -> bf16 elementwise conversion (same layout)
// ---------------------------------------------------------------------------
__global__ void f2bf_kernel(const float* __restrict__ in, __bf16* __restrict__ out, int n) {
    int i = blockIdx.x * blockDim.x + threadIdx.x;
    if (i < n) out[i] = (__bf16)in[i];
}

// ---------------------------------------------------------------------------
// fp32 [K x N] row-major -> bf16 [N x K] row-major (transpose for B fragments)
// ---------------------------------------------------------------------------
__global__ void f2bf_t_kernel(const float* __restrict__ in, __bf16* __restrict__ out,
                              int K, int N) {
    int i = blockIdx.x * blockDim.x + threadIdx.x;
    if (i >= K * N) return;
    int k = i / N, n = i % N;
    out[(size_t)n * K + k] = (__bf16)in[i];
}

// ---------------------------------------------------------------------------
// LayerNorm over rows of 512, output bf16
// ---------------------------------------------------------------------------
__global__ void __launch_bounds__(256, 1)
ln_kernel(const float* __restrict__ x, const float* __restrict__ g,
          const float* __restrict__ be, __bf16* __restrict__ out) {
    int row = blockIdx.x;
    int t = threadIdx.x;                 // 256 threads
    const float* xr = x + (size_t)row * DIM;
    float a = xr[t], b2 = xr[t + 256];
    float s = a + b2, ss = a * a + b2 * b2;
#pragma unroll
    for (int off = 16; off >= 1; off >>= 1) {
        s  += __shfl_xor(s,  off, 32);
        ss += __shfl_xor(ss, off, 32);
    }
    __shared__ float red[16];
    int wave = t >> 5, lane = t & 31;
    if (lane == 0) { red[wave] = s; red[8 + wave] = ss; }
    __syncthreads();
    float st = 0.f, sst = 0.f;
#pragma unroll
    for (int i = 0; i < 8; ++i) { st += red[i]; sst += red[8 + i]; }
    float mean = st * (1.0f / DIM);
    float var  = sst * (1.0f / DIM) - mean * mean;
    float inv  = rsqrtf(var + 1e-5f);
    out[(size_t)row * DIM + t]       = (__bf16)((a  - mean) * inv * g[t]       + be[t]);
    out[(size_t)row * DIM + t + 256] = (__bf16)((b2 - mean) * inv * g[t + 256] + be[t + 256]);
}

// ---------------------------------------------------------------------------
// bf16 GEMM: C[M,N] = A[M,K] (row-major) x B[K,N], with B given TRANSPOSED as
// Bt[N,K] row-major so both fragments load as contiguous 16B-per-lane chunks.
// One wave computes a 32x64 block: 2 A fragments shared across 4 B fragments
// -> 8 WMMAs per K-step.
// MODE 0: bf16 out            MODE 1: bf16 out + bias
// MODE 2: f32 out + bias + residual
// MODE 3: bf16 out transposed for attention V: [(b*DIM+col)*Mkv + key]
// ---------------------------------------------------------------------------
template <int MODE>
__global__ void __launch_bounds__(256, 1)
gemm_bf16_kernel(const __bf16* __restrict__ A,
                 const __bf16* __restrict__ Bt,
                 int M, int N, int K,
                 float* __restrict__ outF,
                 __bf16* __restrict__ outB,
                 const float* __restrict__ bias,
                 const float* __restrict__ residual,
                 int logMkv) {
    int wave = threadIdx.x >> 5;
    int lane = threadIdx.x & 31;
    int stripsN = N >> 6;
    int total = (M >> 5) * stripsN;
    int id = blockIdx.x * (blockDim.x >> 5) + wave;
    if (id >= total) return;                 // wave-uniform: EXEC stays all-1s
    int tm = id / stripsN, tn0 = (id % stripsN) << 2;
    int half = lane >> 4, l16 = lane & 15;
    const __bf16* arow0 = A + (size_t)(tm * 32 + l16) * K;
    const __bf16* arow1 = arow0 + (size_t)16 * K;

    v8f acc[2][4];
    const v8f zero8 = {0.f, 0.f, 0.f, 0.f, 0.f, 0.f, 0.f, 0.f};
#pragma unroll
    for (int mt = 0; mt < 2; ++mt)
#pragma unroll
        for (int t = 0; t < 4; ++t) acc[mt][t] = zero8;

    for (int k0 = 0; k0 < K; k0 += 32) {
        BF16Frag a0, a1;
#pragma unroll
        for (int v = 0; v < 8; ++v) {
            int k = k0 + (half << 3) + ((v >> 2) << 4) + ((v & 3) << 1);
            a0.u[v] = *(const unsigned int*)(arow0 + k);
            a1.u[v] = *(const unsigned int*)(arow1 + k);
        }
#pragma unroll
        for (int t = 0; t < 4; ++t) {
            const __bf16* brow = Bt + (size_t)((tn0 + t) * 16 + l16) * K;
            BF16Frag b;
#pragma unroll
            for (int v = 0; v < 8; ++v) {
                int k = k0 + (half << 3) + ((v >> 2) << 4) + ((v & 3) << 1);
                b.u[v] = *(const unsigned int*)(brow + k);
            }
            acc[0][t] = __builtin_amdgcn_wmma_f32_16x16x32_bf16(false, a0.v, false, b.v,
                                                                (short)0, acc[0][t], false, false);
            acc[1][t] = __builtin_amdgcn_wmma_f32_16x16x32_bf16(false, a1.v, false, b.v,
                                                                (short)0, acc[1][t], false, false);
        }
    }
#pragma unroll
    for (int mt = 0; mt < 2; ++mt) {
#pragma unroll
        for (int t = 0; t < 4; ++t) {
            int col = (tn0 + t) * 16 + l16;
            float bval = (MODE == 1 || MODE == 2) ? bias[col] : 0.f;
#pragma unroll
            for (int v = 0; v < 8; ++v) {
                int row = tm * 32 + mt * 16 + v + (half << 3);
                float val = acc[mt][t][v] + bval;
                if (MODE == 2) {
                    size_t idx = (size_t)row * N + col;
                    outF[idx] = val + residual[idx];
                } else if (MODE == 3) {
                    int bidx = row >> logMkv;
                    int key  = row & ((1 << logMkv) - 1);
                    outB[((size_t)bidx * DIM + col) << logMkv | key] = (__bf16)val;
                } else {
                    outB[(size_t)row * N + col] = (__bf16)val;
                }
            }
        }
    }
}

// ---------------------------------------------------------------------------
// Flash attention: Q [B,4096,512] (head-major cols), K [B,Mkv,512],
// Vt [(b*512+col)*Mkv + key] (transposed so PV B-fragments are packed loads).
// Block = 128 threads = 4 waves; each wave owns 16 query rows of one (b,head).
// launch_bounds(128,1): keep Q frags + 32-VGPR O accumulator resident, no spills.
// ---------------------------------------------------------------------------
__global__ void __launch_bounds__(128, 1)
attn_kernel(const __bf16* __restrict__ Q, const __bf16* __restrict__ Km,
            const __bf16* __restrict__ Vt, __bf16* __restrict__ O,
            int Mkv) {
    const int ld = DIM;
    int wave = threadIdx.x >> 5, lane = threadIdx.x & 31;
    int half = lane >> 4, l16 = lane & 15;
    int b = blockIdx.z, head = blockIdx.y;
    int q0 = blockIdx.x * 64 + wave * 16;

    const __bf16* qp = Q  + ((size_t)b * SEQ_N + q0) * ld + head * D_HEAD;
    const __bf16* kp = Km + (size_t)b * Mkv * ld + head * D_HEAD;
    const __bf16* vp = Vt + ((size_t)b * DIM + head * D_HEAD) * Mkv;

    __shared__ __bf16 lds_p[4][16][32];

    BF16Frag aq0, aq1;
#pragma unroll
    for (int v = 0; v < 8; ++v) {
        int k = (half << 3) + ((v >> 2) << 4) + ((v & 3) << 1);
        aq0.u[v] = *(const unsigned int*)(qp + (size_t)l16 * ld + k);
        aq1.u[v] = *(const unsigned int*)(qp + (size_t)l16 * ld + 32 + k);
    }

    float m_i[8], l_i[8];
    v8f accO[4];
    const v8f zero8 = {0.f, 0.f, 0.f, 0.f, 0.f, 0.f, 0.f, 0.f};
#pragma unroll
    for (int v = 0; v < 8; ++v) { m_i[v] = -__builtin_inff(); l_i[v] = 0.f; }
#pragma unroll
    for (int dt = 0; dt < 4; ++dt) accO[dt] = zero8;

    for (int key0 = 0; key0 < Mkv; key0 += 32) {
        float p[2][8];
#pragma unroll
        for (int jt = 0; jt < 2; ++jt) {
            BF16Frag bk0, bk1;
            const __bf16* krow = kp + (size_t)(key0 + jt * 16 + l16) * ld;
#pragma unroll
            for (int v = 0; v < 8; ++v) {
                int d = (half << 3) + ((v >> 2) << 4) + ((v & 3) << 1);
                bk0.u[v] = *(const unsigned int*)(krow + d);
                bk1.u[v] = *(const unsigned int*)(krow + 32 + d);
            }
            v8f s = zero8;
            s = __builtin_amdgcn_wmma_f32_16x16x32_bf16(false, aq0.v, false, bk0.v,
                                                        (short)0, s, false, false);
            s = __builtin_amdgcn_wmma_f32_16x16x32_bf16(false, aq1.v, false, bk1.v,
                                                        (short)0, s, false, false);
#pragma unroll
            for (int v = 0; v < 8; ++v) p[jt][v] = s[v] * ATTN_SCALE;
        }
        // online softmax over this 32-key block; DPP16 butterflies (rows of 16
        // lanes reduce independently = exactly our half-row ownership)
#pragma unroll
        for (int v = 0; v < 8; ++v) {
            float tmax = red_max16(fmaxf(p[0][v], p[1][v]));
            float mnew  = fmaxf(m_i[v], tmax);
            float scale = __expf(m_i[v] - mnew);
            float p0 = __expf(p[0][v] - mnew);
            float p1 = __expf(p[1][v] - mnew);
            p[0][v] = p0; p[1][v] = p1;
            float rsum = red_sum16(p0 + p1);
            l_i[v] = l_i[v] * scale + rsum;
            m_i[v] = mnew;
#pragma unroll
            for (int dt = 0; dt < 4; ++dt) accO[dt][v] *= scale;
        }
        // stage P through LDS: C-layout -> A-fragment layout
#pragma unroll
        for (int v = 0; v < 8; ++v) {
            lds_p[wave][v + 8 * half][l16]      = (__bf16)p[0][v];
            lds_p[wave][v + 8 * half][16 + l16] = (__bf16)p[1][v];
        }
        asm volatile("s_wait_dscnt 0" ::: "memory");
        BF16Frag ap;
#pragma unroll
        for (int v = 0; v < 8; ++v) {
            int k = (half << 3) + ((v >> 2) << 4) + ((v & 3) << 1);
            ap.u[v] = *(const unsigned int*)(&lds_p[wave][l16][k]);
        }
        // O += P x V over four 16-wide d tiles (packed loads from transposed V)
#pragma unroll
        for (int dt = 0; dt < 4; ++dt) {
            const __bf16* vrow = vp + (size_t)(dt * 16 + l16) * Mkv + key0;
            BF16Frag bv;
#pragma unroll
            for (int v = 0; v < 8; ++v) {
                int kk = (half << 3) + ((v >> 2) << 4) + ((v & 3) << 1);
                bv.u[v] = *(const unsigned int*)(vrow + kk);
            }
            accO[dt] = __builtin_amdgcn_wmma_f32_16x16x32_bf16(false, ap.v, false, bv.v,
                                                               (short)0, accO[dt], false, false);
        }
    }
    __bf16* op = O + ((size_t)b * SEQ_N + q0) * ld + head * D_HEAD;
#pragma unroll
    for (int v = 0; v < 8; ++v) {
        float inv = 1.0f / l_i[v];
        int row = v + 8 * half;
#pragma unroll
        for (int dt = 0; dt < 4; ++dt)
            op[(size_t)row * ld + dt * 16 + l16] = (__bf16)(accO[dt][v] * inv);
    }
}

// ---------------------------------------------------------------------------
// GEGLU: out = val * gelu_exact(gate)   (bf16 in, bf16 out)
// ---------------------------------------------------------------------------
__global__ void geglu_kernel(const __bf16* __restrict__ val, const __bf16* __restrict__ gate,
                             __bf16* __restrict__ out, int n) {
    int i = blockIdx.x * blockDim.x + threadIdx.x;
    if (i >= n) return;
    float v = (float)val[i], g = (float)gate[i];
    float ge = 0.5f * g * (1.0f + erff(g * 0.7071067811865476f));
    out[i] = (__bf16)(v * ge);
}

// ---------------------------------------------------------------------------
// Host-side orchestration
// ---------------------------------------------------------------------------
extern "C" void kernel_launch(void* const* d_in, const int* in_sizes, int n_in,
                              void* d_out, int out_size, void* d_ws, size_t ws_size,
                              hipStream_t stream) {
    const float* x      = (const float*)d_in[0];
    const float* ctx    = (const float*)d_in[1];
    const float* q1_w   = (const float*)d_in[2];
    const float* k1_w   = (const float*)d_in[3];
    const float* v1_w   = (const float*)d_in[4];
    const float* o1_w   = (const float*)d_in[5];
    const float* o1_b   = (const float*)d_in[6];
    const float* q2_w   = (const float*)d_in[7];
    const float* k2_w   = (const float*)d_in[8];
    const float* v2_w   = (const float*)d_in[9];
    const float* o2_w   = (const float*)d_in[10];
    const float* o2_b   = (const float*)d_in[11];
    const float* ffin_w = (const float*)d_in[12];
    const float* ffin_b = (const float*)d_in[13];
    const float* ffout_w= (const float*)d_in[14];
    const float* ffout_b= (const float*)d_in[15];
    const float* ln1_g  = (const float*)d_in[16];
    const float* ln1_b  = (const float*)d_in[17];
    const float* ln2_g  = (const float*)d_in[18];
    const float* ln2_b  = (const float*)d_in[19];
    const float* ln3_g  = (const float*)d_in[20];
    const float* ln3_b  = (const float*)d_in[21];
    float* out = (float*)d_out;

    char* ws = (char*)d_ws;
    size_t off = 0;
    auto alloc = [&](size_t bytes) -> void* {
        void* p = ws + off;
        off = (off + bytes + 255) & ~(size_t)255;
        return p;
    };
    // all weights stored transposed [N x K] bf16
    __bf16* wq1    = (__bf16*)alloc((size_t)DIM * DIM * 2);
    __bf16* wk1    = (__bf16*)alloc((size_t)DIM * DIM * 2);
    __bf16* wv1    = (__bf16*)alloc((size_t)DIM * DIM * 2);
    __bf16* wo1    = (__bf16*)alloc((size_t)DIM * DIM * 2);
    __bf16* wq2    = (__bf16*)alloc((size_t)DIM * DIM * 2);
    __bf16* wk2    = (__bf16*)alloc((size_t)CTX_DIM * DIM * 2);
    __bf16* wv2    = (__bf16*)alloc((size_t)CTX_DIM * DIM * 2);
    __bf16* wo2    = (__bf16*)alloc((size_t)DIM * DIM * 2);
    __bf16* wffin  = (__bf16*)alloc((size_t)DIM * 2 * FF_INNER * 2);
    __bf16* wffout = (__bf16*)alloc((size_t)FF_INNER * DIM * 2);
    __bf16* hbuf   = (__bf16*)alloc((size_t)ROWS * DIM * 2);
    __bf16* ctxb   = (__bf16*)alloc((size_t)CROWS * CTX_DIM * 2);
    __bf16* qb     = (__bf16*)alloc((size_t)ROWS * DIM * 2);
    __bf16* kb     = (__bf16*)alloc((size_t)ROWS * DIM * 2);
    __bf16* vb     = (__bf16*)alloc((size_t)ROWS * DIM * 2);   // transposed V
    __bf16* aob    = (__bf16*)alloc((size_t)ROWS * DIM * 2);
    float*  xbuf   = (float*) alloc((size_t)ROWS * DIM * 4);
    __bf16* valb   = (__bf16*)alloc((size_t)ROWS * FF_INNER * 2);
    __bf16* gateb  = (__bf16*)alloc((size_t)ROWS * FF_INNER * 2);
    __bf16* geglub = (__bf16*)alloc((size_t)ROWS * FF_INNER * 2);

    auto convT = [&](const float* src, __bf16* dst, int K, int N) {
        int n = K * N;
        f2bf_t_kernel<<<(n + 255) / 256, 256, 0, stream>>>(src, dst, K, N);
    };
    convT(q1_w,    wq1,    DIM, DIM);
    convT(k1_w,    wk1,    DIM, DIM);
    convT(v1_w,    wv1,    DIM, DIM);
    convT(o1_w,    wo1,    DIM, DIM);
    convT(q2_w,    wq2,    DIM, DIM);
    convT(k2_w,    wk2,    CTX_DIM, DIM);
    convT(v2_w,    wv2,    CTX_DIM, DIM);
    convT(o2_w,    wo2,    DIM, DIM);
    convT(ffin_w,  wffin,  DIM, 2 * FF_INNER);
    convT(ffout_w, wffout, FF_INNER, DIM);
    {
        int n = CROWS * CTX_DIM;
        f2bf_kernel<<<(n + 255) / 256, 256, 0, stream>>>(ctx, ctxb, n);
    }

    auto launch_gemm = [&](auto kern, const __bf16* A, const __bf16* Bt,
                           int M, int N, int K, float* oF, __bf16* oB,
                           const float* bias, const float* res, int logMkv) {
        int strips = (M >> 5) * (N >> 6);
        kern<<<(strips + 7) / 8, 256, 0, stream>>>(A, Bt, M, N, K, oF, oB, bias, res, logMkv);
    };

    // ---- self-attention block ----
    ln_kernel<<<ROWS, 256, 0, stream>>>(x, ln1_g, ln1_b, hbuf);
    launch_gemm(gemm_bf16_kernel<0>, hbuf, wq1, ROWS, DIM, DIM, nullptr, qb, nullptr, nullptr, 0);
    launch_gemm(gemm_bf16_kernel<0>, hbuf, wk1, ROWS, DIM, DIM, nullptr, kb, nullptr, nullptr, 0);
    launch_gemm(gemm_bf16_kernel<3>, hbuf, wv1, ROWS, DIM, DIM, nullptr, vb, nullptr, nullptr, 12);
    attn_kernel<<<dim3(SEQ_N / 64, HEADS, BATCH), 128, 0, stream>>>(qb, kb, vb, aob, SEQ_N);
    launch_gemm(gemm_bf16_kernel<2>, aob, wo1, ROWS, DIM, DIM, out, nullptr, o1_b, x, 0);

    // ---- cross-attention block ----
    ln_kernel<<<ROWS, 256, 0, stream>>>(out, ln2_g, ln2_b, hbuf);
    launch_gemm(gemm_bf16_kernel<0>, hbuf, wq2, ROWS, DIM, DIM, nullptr, qb, nullptr, nullptr, 0);
    launch_gemm(gemm_bf16_kernel<0>, ctxb, wk2, CROWS, DIM, CTX_DIM, nullptr, kb, nullptr, nullptr, 0);
    launch_gemm(gemm_bf16_kernel<3>, ctxb, wv2, CROWS, DIM, CTX_DIM, nullptr, vb, nullptr, nullptr, 8);
    attn_kernel<<<dim3(SEQ_N / 64, HEADS, BATCH), 128, 0, stream>>>(qb, kb, vb, aob, CTX_M);
    launch_gemm(gemm_bf16_kernel<2>, aob, wo2, ROWS, DIM, DIM, xbuf, nullptr, o2_b, out, 0);

    // ---- GEGLU feed-forward block ----
    ln_kernel<<<ROWS, 256, 0, stream>>>(xbuf, ln3_g, ln3_b, hbuf);
    launch_gemm(gemm_bf16_kernel<1>, hbuf, wffin, ROWS, FF_INNER, DIM,
                nullptr, valb, ffin_b, nullptr, 0);
    launch_gemm(gemm_bf16_kernel<1>, hbuf, wffin + (size_t)FF_INNER * DIM, ROWS, FF_INNER, DIM,
                nullptr, gateb, ffin_b + FF_INNER, nullptr, 0);
    {
        int n = ROWS * FF_INNER;
        geglu_kernel<<<(n + 255) / 256, 256, 0, stream>>>(valb, gateb, geglub, n);
    }
    launch_gemm(gemm_bf16_kernel<2>, geglub, wffout, ROWS, DIM, FF_INNER, out, nullptr,
                ffout_b, xbuf, 0);
}
